// QCNNHybridModel_65481071406518
// MI455X (gfx1250) — matrix operand order repaired and checked by reading, not compiled, not run on hardware
//
#include <hip/hip_runtime.h>
#include <hip/hip_bf16.h>

typedef __attribute__((ext_vector_type(8)))  float  v8f;
typedef __attribute__((ext_vector_type(16))) __bf16 v16bf;
typedef __attribute__((ext_vector_type(8)))  __bf16 v8bf;

#define D_K     1024   // inner dim (== R for layer0 input)
#define R_COLS  1024   // gram columns
#define TILE_M  64     // B rows per workgroup
#define CHUNK_N 128    // gram columns processed per chunk
#define KB      64     // K-block staged per async round (2 WMMA K-steps)

// ---------------------------------------------------------------------------
// Prep: convert ref ([R, D] f32, row-major) to bf16 (workspace) and compute
// r2[j] = sum_k ref_bf16[j,k]^2.  ref row j is exactly WMMA B-matrix column j
// with K contiguous, so staging keeps K-major layout.
// ---------------------------------------------------------------------------
__global__ __launch_bounds__(256) void rbf_prep_kernel(
    const float* __restrict__ ref, __bf16* __restrict__ refb,
    float* __restrict__ r2) {
  __shared__ float part[256];
  const int j = blockIdx.x;
  const int tid = threadIdx.x;
  float s = 0.0f;
#pragma unroll
  for (int i = 0; i < D_K / 256; ++i) {
    const int k = tid + i * 256;
    const float v = ref[(size_t)j * D_K + k];
    const __bf16 b = (__bf16)v;
    refb[(size_t)j * D_K + k] = b;
    const float vb = (float)b;
    s += vb * vb;
  }
  part[tid] = s;
  __syncthreads();
  for (int off = 128; off > 0; off >>= 1) {
    if (tid < off) part[tid] += part[tid + off];
    __syncthreads();
  }
  if (tid == 0) r2[j] = part[0];
}

// ---------------------------------------------------------------------------
// Fused: gram tile (bf16 WMMA, f32 acc) -> exp(-d2) -> @W0 -> tanh -> tiny MLP
// Grid: B/TILE_M workgroups of 256 threads (8 waves: 4 row-groups x 2 halves).
// ref K-blocks staged once per WG via async global->LDS (SADDR form, 32-bit
// per-lane offsets), double-buffered against the WMMA consume loop.
// ---------------------------------------------------------------------------
__global__ __launch_bounds__(256) void rbf_mlp_kernel(
    const float* __restrict__ x, const __bf16* __restrict__ refb,
    const float* __restrict__ r2g,
    const float* __restrict__ W0, const float* __restrict__ b0,
    const float* __restrict__ W1, const float* __restrict__ b1,
    const float* __restrict__ W2, const float* __restrict__ b2,
    const float* __restrict__ W3, const float* __restrict__ b3,
    const float* __restrict__ W4, const float* __restrict__ b4,
    const float* __restrict__ W5, const float* __restrict__ b5,
    float* __restrict__ out) {
  __shared__ __bf16 xb[TILE_M * D_K];           // 128 KB: x tile, bf16
  __shared__ __bf16 rbuf[2][CHUNK_N * KB];      //  32 KB: ref K-block, dbl-buf
  __shared__ __bf16 Sb[TILE_M * CHUNK_N];       //  16 KB: exp(-d2) chunk
  __shared__ float  x2s[TILE_M];
  __shared__ float  h0s[TILE_M * 16];

  const int tid  = threadIdx.x;
  const int lane = tid & 31;
  const int wave = tid >> 5;
  const int mg   = wave >> 1;          // row-group 0..3 (16 rows each)
  const int ch   = wave & 1;           // column half 0..1 (64 cols each)
  const int hs   = lane >> 4;          // half-select within wave32
  const int ln   = lane & 15;
  const size_t rowBase = (size_t)blockIdx.x * TILE_M;

  // LDS byte addresses of the two staging buffers (scalars, no arrays)
  const unsigned rl0 = (unsigned)(uintptr_t)&rbuf[0][0];
  const unsigned rl1 = (unsigned)(uintptr_t)&rbuf[1][0];

  // Per-thread static byte offsets of this thread's 4 staged 16B segments.
  unsigned goff[4], loff[4];
#pragma unroll
  for (int i = 0; i < 4; ++i) {
    const int s  = tid + i * 256;      // 16-byte segment id (1024 total)
    const int c  = s >> 3;             // column within chunk
    const int kk = (s & 7) * 8;        // K offset within block
    goff[i] = (unsigned)(c * (D_K * 2) + kk * 2);
    loff[i] = (unsigned)((c * KB + kk) * 2);
  }

  // Issue this thread's share of one ref K-block: async global->LDS b128,
  // GVS addressing (SGPR base = refb, 32-bit VGPR byte offset).
  auto stage = [&](unsigned gbyte, unsigned lbase) {
#pragma unroll
    for (int i = 0; i < 4; ++i) {
      asm volatile("global_load_async_to_lds_b128 %0, %1, %2"
                   :: "v"(lbase + loff[i]), "v"(gbyte + goff[i]), "s"(refb)
                   : "memory");
    }
  };

  // ---- Stage x tile into LDS as bf16 (f32 read from HBM exactly once) ----
#pragma unroll 4
  for (int i = 0; i < (TILE_M * D_K) / (256 * 4); ++i) {
    const int idx4 = tid + i * 256;
    const int r = idx4 >> 8;                 // (idx4*4) / 1024
    const int k = (idx4 << 2) & (D_K - 1);
    const float4 v = *(const float4*)(x + (rowBase + r) * D_K + k);
    xb[r * D_K + k + 0] = (__bf16)v.x;
    xb[r * D_K + k + 1] = (__bf16)v.y;
    xb[r * D_K + k + 2] = (__bf16)v.z;
    xb[r * D_K + k + 3] = (__bf16)v.w;
  }
  if (tid < TILE_M) x2s[tid] = 0.0f;
  __syncthreads();

  // ---- x2[r] = sum_k xb[r,k]^2 (4 threads/row, ds_add_f32 reduce) ----
  {
    const int r = tid >> 2, seg = tid & 3;
    float s = 0.0f;
#pragma unroll 8
    for (int k = 0; k < 256; ++k) {
      const float v = (float)xb[r * D_K + seg * 256 + k];
      s += v * v;
    }
    atomicAdd(&x2s[r], s);
  }
  __syncthreads();

  float hacc[4] = {0.0f, 0.0f, 0.0f, 0.0f};  // layer-0 partials, [4 rows][f]
  const int f  = tid & 15;                   // feature owned in layer-0 reduce
  const int r0 = tid >> 4;

#pragma unroll 1
  for (int rc = 0; rc < R_COLS / CHUNK_N; ++rc) {
    const int colBase = rc * CHUNK_N;
    const unsigned chunkByte = (unsigned)(colBase * (D_K * 2));
    v8f acc[4] = {v8f{}, v8f{}, v8f{}, v8f{}};

    stage(chunkByte, rl0);  // prologue: stage K-block 0

#pragma unroll 1
    for (int blk = 0; blk < D_K / KB; ++blk) {
      // Own async copies done -> barrier makes all waves' staging visible.
      asm volatile("s_wait_asynccnt 0x0" ::: "memory");
      __syncthreads();
      if (blk + 1 < D_K / KB)
        stage(chunkByte + (unsigned)((blk + 1) * KB * 2),
              ((blk + 1) & 1) ? rl1 : rl0);

      const __bf16* rb = (blk & 1) ? &rbuf[1][0] : &rbuf[0][0];
#pragma unroll
      for (int ks = 0; ks < KB; ks += 32) {
        const int kb = blk * KB + ks;
        // A fragment: lane holds 16 contiguous K values (matches B)
        const int arow = mg * 16 + ln;
        const int k0 = kb + hs * 16;
        const v8bf alo = *(const v8bf*)&xb[arow * D_K + k0];
        const v8bf ahi = *(const v8bf*)&xb[arow * D_K + k0 + 8];
        const v16bf a = __builtin_shufflevector(alo, ahi,
            0, 1, 2, 3, 4, 5, 6, 7, 8, 9, 10, 11, 12, 13, 14, 15);
#pragma unroll
        for (int t = 0; t < 4; ++t) {
          const int cc = ch * 64 + t * 16 + ln;
          // lane holds 16 contiguous K values of its B column
          const v16bf b = *(const v16bf*)&rb[cc * KB + ks + hs * 16];
          acc[t] = __builtin_amdgcn_wmma_f32_16x16x32_bf16(
              false, a, false, b, (short)0, acc[t], false, false);
        }
      }
    }

    // ---- exp(-max(x2 + r2 - 2*dot, 0)) -> Sb (bf16) ----
    float x2v[8];
#pragma unroll
    for (int i = 0; i < 8; ++i) x2v[i] = x2s[mg * 16 + i + 8 * hs];
#pragma unroll
    for (int t = 0; t < 4; ++t) {
      const int cc  = ch * 64 + t * 16 + ln;
      const float rv = r2g[colBase + cc];
#pragma unroll
      for (int i = 0; i < 8; ++i) {
        const int m = mg * 16 + i + 8 * hs;
        float d = x2v[i] + rv - 2.0f * acc[t][i];
        d = fmaxf(d, 0.0f);
        Sb[m * CHUNK_N + cc] = (__bf16)__expf(-d);
      }
    }
    __syncthreads();

    // ---- layer-0 partial reduce: hacc[r, f] += S[r, k] * W0[colBase+k, f] ----
    for (int k = 0; k < CHUNK_N; ++k) {
      const float w = W0[(size_t)(colBase + k) * 16 + f];
#pragma unroll
      for (int j = 0; j < 4; ++j)
        hacc[j] += (float)Sb[(r0 + 16 * j) * CHUNK_N + k] * w;
    }
    __syncthreads();  // protect Sb before next chunk overwrites it
  }

  // ---- bias + tanh, share layer-0 activations ----
#pragma unroll
  for (int j = 0; j < 4; ++j)
    h0s[(r0 + 16 * j) * 16 + f] = tanhf(hacc[j] + b0[f]);
  __syncthreads();

  // ---- tiny MLP tail: 16 -> 12 -> 8 -> 4 -> 4 -> 1 (one thread per row) ----
  if (tid < TILE_M) {
    float h[16];
#pragma unroll
    for (int k = 0; k < 16; ++k) h[k] = h0s[tid * 16 + k];
    float t1[12];
#pragma unroll
    for (int j = 0; j < 12; ++j) {
      float s = b1[j];
#pragma unroll
      for (int k = 0; k < 16; ++k) s += h[k] * W1[k * 12 + j];
      t1[j] = tanhf(s);
    }
    float t2[8];
#pragma unroll
    for (int j = 0; j < 8; ++j) {
      float s = b2[j];
#pragma unroll
      for (int k = 0; k < 12; ++k) s += t1[k] * W2[k * 8 + j];
      t2[j] = tanhf(s);
    }
    float t3[4];
#pragma unroll
    for (int j = 0; j < 4; ++j) {
      float s = b3[j];
#pragma unroll
      for (int k = 0; k < 8; ++k) s += t2[k] * W3[k * 4 + j];
      t3[j] = tanhf(s);
    }
    float t4[4];
#pragma unroll
    for (int j = 0; j < 4; ++j) {
      float s = b4[j];
#pragma unroll
      for (int k = 0; k < 4; ++k) s += t3[k] * W4[k * 4 + j];
      t4[j] = tanhf(s);
    }
    float s5 = b5[0];
#pragma unroll
    for (int k = 0; k < 4; ++k) s5 += t4[k] * W5[k];
    out[rowBase + tid] = 1.0f / (1.0f + __expf(-s5));
  }
}

extern "C" void kernel_launch(void* const* d_in, const int* in_sizes, int n_in,
                              void* d_out, int out_size, void* d_ws, size_t ws_size,
                              hipStream_t stream) {
  const float* x   = (const float*)d_in[0];
  const float* ref = (const float*)d_in[1];
  const float* W0 = (const float*)d_in[2];
  const float* b0 = (const float*)d_in[3];
  const float* W1 = (const float*)d_in[4];
  const float* b1 = (const float*)d_in[5];
  const float* W2 = (const float*)d_in[6];
  const float* b2 = (const float*)d_in[7];
  const float* W3 = (const float*)d_in[8];
  const float* b3 = (const float*)d_in[9];
  const float* W4 = (const float*)d_in[10];
  const float* b4 = (const float*)d_in[11];
  const float* W5 = (const float*)d_in[12];
  const float* b5 = (const float*)d_in[13];
  float* out = (float*)d_out;

  // workspace: [0, 4KB) r2, [4KB, 4KB + 2MB) ref in bf16
  float*  r2   = (float*)d_ws;
  __bf16* refb = (__bf16*)((char*)d_ws + 4096);

  rbf_prep_kernel<<<R_COLS, 256, 0, stream>>>(ref, refb, r2);
  rbf_mlp_kernel<<<32768 / TILE_M, 256, 0, stream>>>(
      x, refb, r2, W0, b0, W1, b1, W2, b2, W3, b3, W4, b4, W5, b5, out);
}
